// LSTMClassifier_39805756899517
// MI455X (gfx1250) — matrix operand order repaired and checked by reading, not compiled, not run on hardware
//
#include <hip/hip_runtime.h>
#include <stdint.h>

// ---------------------------------------------------------------------------
// LSTM classifier for MI455X (gfx1250, wave32, WMMA).
// B=256, T=512, E=256, H=512, concat K=768, gates N=4*512.
// Strategy:
//   * Persistent kernel: 32 blocks x 256 threads (8 waves). Block nb owns
//     hidden columns [nb*16, nb*16+16) for all 4 gates. Wave w owns batch
//     rows [w*32, w*32+32) as two 16x16 M-tiles. Cell state c stays in VGPRs.
//   * Weights pre-converted to bf16 and pre-blocked [nb][gate][16][768]; each
//     block DMAs its 96KB slice into LDS once via the Tensor Data Mover.
//   * G=[x_t|h] double-buffered bf16 [2][256][768] in workspace; one grid
//     barrier per step (monotonic atomic counter, re-zeroed each call).
// ---------------------------------------------------------------------------

#define Bsz   256
#define Tlen  512
#define Esz   256
#define Hsz   512
#define Kc    768           // E + H
#define NBLK  32
#define NTHR  256
#define WSLICE (4*16*Kc)    // bf16 elems per block weight slice (49152)

typedef __attribute__((ext_vector_type(16))) __bf16 v16bf;
typedef __attribute__((ext_vector_type(8)))  __bf16 v8bf;
typedef __attribute__((ext_vector_type(8)))  float  v8f;
typedef __attribute__((ext_vector_type(4))) unsigned int u32x4;
typedef __attribute__((ext_vector_type(8)))  int    i32x8;
typedef __attribute__((ext_vector_type(4)))  int    i32x4;

// workspace byte offsets
#define OFF_WT2 0u              // 32*4*16*768*2 = 3,145,728
#define OFF_EMB 3145728u        // 14*256*2      = 7,168 (padded)
#define OFF_G   3153920u        // 2*256*768*2   = 786,432
#define OFF_LH  3940352u        // 256*512*4     = 524,288
#define OFF_CNT 4464640u        // barrier counter

__device__ __forceinline__ float sigf(float x) { return 1.0f / (1.0f + __expf(-x)); }

__device__ __forceinline__ v16bf load_a16(const __bf16* p) {
    v8bf lo = *(const v8bf*)p;          // K .. K+7   (this lane's low half)
    v8bf hi = *(const v8bf*)(p + 8);    // K+8..K+15 (this lane's high half)
    v16bf r;
#pragma unroll
    for (int i = 0; i < 8; ++i) { r[i] = lo[i]; r[i + 8] = hi[i]; }
    return r;
}

__device__ __forceinline__ void grid_barrier(unsigned* cnt, unsigned target) {
    __threadfence();
    __syncthreads();
    if (threadIdx.x == 0) {
        atomicAdd(cnt, 1u);
        while (__hip_atomic_load(cnt, __ATOMIC_ACQUIRE, __HIP_MEMORY_SCOPE_AGENT) < target)
            __builtin_amdgcn_s_sleep(1);
    }
    __syncthreads();
}

// --- prep: weights f32 [768,512] (k-major) -> bf16 blocked [nb][g][16][768] ---
__global__ void prep_weights(const float* __restrict__ Wf, const float* __restrict__ Wi,
                             const float* __restrict__ Wc, const float* __restrict__ Wo,
                             unsigned short* __restrict__ wt2_us) {
    __bf16* wt2 = (__bf16*)wt2_us;
    int idx = blockIdx.x * blockDim.x + threadIdx.x;   // [0, 4*512*768)
    if (idx >= 4 * Hsz * Kc) return;
    int k = idx % Kc;
    int n = (idx / Kc) % Hsz;
    int g = idx / (Kc * Hsz);
    const float* W = (g == 0) ? Wf : (g == 1) ? Wi : (g == 2) ? Wc : Wo;
    int nb = n >> 4, nl = n & 15;
    wt2[((size_t)(nb * 4 + g) * 16 + nl) * Kc + k] = (__bf16)W[(size_t)k * Hsz + n];
}

// --- prep: emb -> bf16, zero barrier counter ---
__global__ void prep_misc(const float* __restrict__ emb, unsigned short* __restrict__ emb_us,
                          unsigned* __restrict__ cnt) {
    int gid = blockIdx.x * blockDim.x + threadIdx.x;
    if (gid < 14 * Esz) ((__bf16*)emb_us)[gid] = (__bf16)emb[gid];
    if (gid == 4095) *cnt = 0u;
}

// --- persistent LSTM recurrence ---
__global__ void lstm_persist(const unsigned short* __restrict__ wt2_us,
                             const unsigned short* __restrict__ emb_us,
                             unsigned short* __restrict__ g_us,
                             float* __restrict__ lasth,
                             unsigned* __restrict__ cnt,
                             const int* __restrict__ x, const int* __restrict__ lens,
                             const float* __restrict__ bfv, const float* __restrict__ biv,
                             const float* __restrict__ bcv, const float* __restrict__ bov) {
    extern __shared__ __align__(32) unsigned short smem_us[];
    __bf16* wlds = (__bf16*)smem_us;                      // [4][16][768] bf16 = 96KB

    const int nb   = blockIdx.x;
    const int tid  = threadIdx.x;
    const int lane = tid & 31;
    const int wv   = tid >> 5;
    const int colL = lane & 15;
    const int hiL  = lane >> 4;

    const __bf16* wt2  = (const __bf16*)wt2_us;
    const __bf16* embb = (const __bf16*)emb_us;
    __bf16* gb0 = (__bf16*)g_us;                          // buffer 0: [256][768]
    __bf16* gb1 = gb0 + (size_t)Bsz * Kc;                 // buffer 1

    // ---- TDM: DMA this block's contiguous 96KB weight slice into LDS ----
    if (wv == 0) {
        const __bf16* wsrc = wt2 + (size_t)nb * WSLICE;
        unsigned lds_addr  = (unsigned)(uintptr_t)wlds;   // low 32 bits = LDS offset
        unsigned long long ga = (unsigned long long)(uintptr_t)wsrc;
        u32x4 g0;
        g0[0] = 1u;                                       // count = 1 valid descriptor
        g0[1] = lds_addr;                                 // lds_addr
        g0[2] = (unsigned)(ga & 0xffffffffu);             // global_addr[31:0]
        g0[3] = (unsigned)(((ga >> 32) & 0x01ffffffu) | (2u << 30)); // addr hi | type=2
        const unsigned elems = (unsigned)WSLICE;          // 1-D tile, 2B elements
        i32x8 g1;
        g1[0] = (int)(1u << 16);                          // data_size = 1 (2 bytes)
        g1[1] = (int)((elems & 0xffffu) << 16);           // tensor_dim0[15:0]
        g1[2] = (int)((elems >> 16) | (1u << 16));        // tensor_dim0[31:16] | tensor_dim1=1
        g1[3] = (int)((elems & 0xffffu) << 16);           // tile_dim0
        g1[4] = 0;                                        // tile_dim1/2 unused
        g1[5] = (int)elems;                               // tensor_dim0_stride lo
        g1[6] = 0;
        g1[7] = 0;
        i32x4 z4; z4[0] = 0; z4[1] = 0; z4[2] = 0; z4[3] = 0;
        i32x8 z8;
#pragma unroll
        for (int i = 0; i < 8; ++i) z8[i] = 0;
        // clang-23 / therock signature: (u32x4, i32x8, i32x4, i32x4, i32x8, i32 cpol)
        __builtin_amdgcn_tensor_load_to_lds(g0, g1, z4, z4, z8, 0);
        __builtin_amdgcn_s_wait_tensorcnt(0);
    }
    __syncthreads();

    // per-lane constants
    const int hcol = nb * 16 + colL;
    const float vbf = bfv[hcol], vbi = biv[hcol], vbc = bcv[hcol], vbo = bov[hcol];
    const int mt0 = wv * 2;                               // this wave's first M-tile

    float cst[2][8];
#pragma unroll
    for (int m = 0; m < 2; ++m)
#pragma unroll
        for (int r = 0; r < 8; ++r) cst[m][r] = 0.0f;

    // ---- stage step-0 inputs into buffer 0 ----
    {
        int row = nb * 8 + (tid >> 5);                    // x part: 8 rows per block
        int c8  = (tid & 31) * 8;
        int tok = x[(size_t)row * Tlen + 0];
        *(v8bf*)(gb0 + (size_t)row * Kc + c8) = *(const v8bf*)(embb + (size_t)tok * Esz + c8);
        // h part (cols 256 + [nb*16, +16)) = 0 for all 256 rows; thread tid -> row tid
        v8bf z8v = {};
        __bf16* hz = gb0 + (size_t)tid * Kc + Esz + nb * 16;
        *(v8bf*)hz = z8v;
        *(v8bf*)(hz + 8) = z8v;
    }
    grid_barrier(cnt, NBLK * 1u);

    // ---- time loop ----
    for (int t = 0; t < Tlen; ++t) {
        const __bf16* gc = (t & 1) ? gb1 : gb0;
        __bf16* gn       = (t & 1) ? gb0 : gb1;

        v8f acc[2][4];
#pragma unroll
        for (int m = 0; m < 2; ++m)
#pragma unroll
            for (int g = 0; g < 4; ++g) acc[m][g] = (v8f){};

        const __bf16* arow0 = gc + (size_t)(mt0 * 16 + colL) * Kc + hiL * 16;
        const __bf16* arow1 = arow0 + 16 * Kc;

#pragma unroll
        for (int kk = 0; kk < Kc / 32; ++kk) {            // 24 K-iterations
            int kb = kk * 32;
            v16bf a0 = load_a16(arow0 + kb);
            v16bf a1 = load_a16(arow1 + kb);
#pragma unroll
            for (int g = 0; g < 4; ++g) {
                const __bf16* bp = wlds + (size_t)(g * 16 + colL) * Kc + kb + hiL * 16;
                v16bf b = *(const v16bf*)bp;              // contiguous 32B from LDS
                acc[0][g] = __builtin_amdgcn_wmma_f32_16x16x32_bf16(
                    false, a0, false, b, (short)0, acc[0][g], false, false);
                acc[1][g] = __builtin_amdgcn_wmma_f32_16x16x32_bf16(
                    false, a1, false, b, (short)0, acc[1][g], false, false);
            }
        }

        // ---- cell update entirely in registers ----
#pragma unroll
        for (int m = 0; m < 2; ++m) {
            int rbase = (mt0 + m) * 16 + hiL * 8;
#pragma unroll
            for (int r = 0; r < 8; ++r) {
                int row  = rbase + r;
                float fg = sigf(acc[m][0][r] + vbf);
                float ig = sigf(acc[m][1][r] + vbi);
                float cg = tanhf(acc[m][2][r] + vbc);
                float og = sigf(acc[m][3][r] + vbo);
                float cn = fg * cst[m][r] + ig * cg;
                cst[m][r] = cn;
                float hn = og * tanhf(cn);
                gn[(size_t)row * Kc + Esz + hcol] = (__bf16)hn;
                if (lens[row] == t + 1) lasth[(size_t)row * Hsz + hcol] = hn;
            }
        }

        // ---- stage x_{t+1} into the next buffer ----
        if (t + 1 < Tlen) {
            int row = nb * 8 + (tid >> 5);
            int c8  = (tid & 31) * 8;
            int tok = x[(size_t)row * Tlen + (t + 1)];
            *(v8bf*)(gn + (size_t)row * Kc + c8) = *(const v8bf*)(embb + (size_t)tok * Esz + c8);
        }

        grid_barrier(cnt, (unsigned)(NBLK * (t + 2)));
    }
}

// --- final fc + sigmoid: out[b] = sigmoid(lasth[b,:] . fcW + fcb) ---
__global__ void fc_kernel(const float* __restrict__ lasth, const float* __restrict__ fcW,
                          const float* __restrict__ fcb, float* __restrict__ out) {
    int b = blockIdx.x;
    float s = 0.0f;
    for (int h = threadIdx.x; h < Hsz; h += 64) s += lasth[(size_t)b * Hsz + h] * fcW[h];
#pragma unroll
    for (int off = 16; off > 0; off >>= 1) s += __shfl_down(s, off, 32);
    __shared__ float part[2];
    if ((threadIdx.x & 31) == 0) part[threadIdx.x >> 5] = s;
    __syncthreads();
    if (threadIdx.x == 0) out[b] = sigf(part[0] + part[1] + fcb[0]);
}

extern "C" void kernel_launch(void* const* d_in, const int* in_sizes, int n_in,
                              void* d_out, int out_size, void* d_ws, size_t ws_size,
                              hipStream_t stream) {
    (void)in_sizes; (void)n_in; (void)out_size; (void)ws_size;
    const int*   x    = (const int*)d_in[0];
    const int*   lens = (const int*)d_in[1];
    const float* emb  = (const float*)d_in[2];
    const float* Wf   = (const float*)d_in[3];
    const float* bf_  = (const float*)d_in[4];
    const float* Wi   = (const float*)d_in[5];
    const float* bi_  = (const float*)d_in[6];
    const float* Wc   = (const float*)d_in[7];
    const float* bc_  = (const float*)d_in[8];
    const float* Wo   = (const float*)d_in[9];
    const float* bo_  = (const float*)d_in[10];
    const float* fcW  = (const float*)d_in[11];
    const float* fcb  = (const float*)d_in[12];

    char* ws = (char*)d_ws;
    unsigned short* wt2  = (unsigned short*)(ws + OFF_WT2);
    unsigned short* embb = (unsigned short*)(ws + OFF_EMB);
    unsigned short* gbuf = (unsigned short*)(ws + OFF_G);
    float*          lh   = (float*)(ws + OFF_LH);
    unsigned*       cnt  = (unsigned*)(ws + OFF_CNT);

    prep_weights<<<(4 * Hsz * Kc + NTHR - 1) / NTHR, NTHR, 0, stream>>>(Wf, Wi, Wc, Wo, wt2);
    prep_misc<<<16, NTHR, 0, stream>>>(emb, embb, cnt);
    lstm_persist<<<NBLK, NTHR, WSLICE * 2, stream>>>(wt2, embb, gbuf, lh, cnt,
                                                     x, lens, bf_, bi_, bc_, bo_);
    fc_kernel<<<Bsz, 64, 0, stream>>>(lh, fcW, fcb, (float*)d_out);
}
